// SlotAttentionInvTransScale_58076547776909
// MI455X (gfx1250) — matrix operand order repaired and checked by reading, not compiled, not run on hardware
//
#include <hip/hip_runtime.h>
#include <hip/hip_bf16.h>
#include <math.h>

#define B_  8
#define N_  1024
#define K_  8
#define D_  256
#define H_  512
#define BK_ 64
#define EPS_   1e-8f
#define LNEPS_ 1e-5f
#define SCALE_ 5.0f

typedef __bf16 bf16_t;
typedef __attribute__((ext_vector_type(16))) __bf16 bf16x16;
typedef __attribute__((ext_vector_type(8)))  float  f32x8;

struct Q2  { uint4 a, b; };      // 32B = 16 bf16 (one WMMA A/B fragment)
struct BF8 { bf16_t e[8]; };     // 16B = 8 bf16

__device__ __forceinline__ float gridX(int n){ return -1.0f + (2.0f/31.0f)*(float)(n>>5); }
__device__ __forceinline__ float gridY(int n){ return -1.0f + (2.0f/31.0f)*(float)(n&31); }

__device__ __forceinline__ float blockSum256(float v, float* buf){
  int t = threadIdx.x;
  buf[t] = v; __syncthreads();
  for (int s = 128; s > 0; s >>= 1){ if (t < s) buf[t] += buf[t+s]; __syncthreads(); }
  float r = buf[0]; __syncthreads(); return r;
}

// ---------------- P0: x = LN(inputs) -> bf16 ----------------
__global__ void k_ln_inputs(const float* __restrict__ inp, const float* __restrict__ g,
                            const float* __restrict__ b, bf16_t* __restrict__ xq){
  __shared__ float buf[256];
  int m = blockIdx.x, t = threadIdx.x;
  float x = inp[(size_t)m*D_ + t];
  float mean = blockSum256(x, buf) * (1.0f/D_);
  float d = x - mean;
  float var = blockSum256(d*d, buf) * (1.0f/D_);
  float xn = d * rsqrtf(var + LNEPS_) * g[t] + b[t];
  xq[(size_t)m*D_ + t] = (bf16_t)xn;
}

// ---------------- P1: pack weight into WMMA B-fragment order ----------------
// out[((kk*NT+ct)*32 + lane)*16 + 2j+p] = W[kk*32 + (j>>2)*16 + half*8 + 2(j&3) + p][ct*16 + lane%16]
__global__ void k_packB(const float* __restrict__ W, bf16_t* __restrict__ out, int NT){
  int ct = blockIdx.x % NT, kk = blockIdx.x / NT;
  int lane = threadIdx.x, half = lane >> 4, col = ct*16 + (lane & 15);
  int Ncols = NT*16;
  size_t base = ((size_t)blockIdx.x*32 + lane)*16;
  #pragma unroll
  for (int j = 0; j < 8; ++j)
    #pragma unroll
    for (int p = 0; p < 2; ++p){
      int kg = kk*32 + (j>>2)*16 + half*8 + 2*(j&3) + p;
      out[base + j*2 + p] = (bf16_t)W[(size_t)kg*Ncols + col];
    }
}

// ---------------- G1: k = x@wk, v = x@wv (WMMA bf16, f32 out) ----------------
__global__ __launch_bounds__(128) void k_proj(const bf16_t* __restrict__ xq,
                       const bf16_t* __restrict__ wkP, const bf16_t* __restrict__ wvP,
                       float* __restrict__ kout, float* __restrict__ vout){
  int wid  = blockIdx.x*4 + (threadIdx.x >> 5);
  int lane = threadIdx.x & 31, half = lane >> 4, r = lane & 15;
  int which = wid >> 13;           // 0 = k, 1 = v   (8192 tiles each)
  int tile  = wid & 8191;
  int mt = tile >> 4, nt = tile & 15;
  const bf16_t* wp = which ? wvP : wkP;
  float* outp      = which ? vout : kout;
  const bf16_t* arow = xq + ((size_t)(mt*16 + r))*D_;
  f32x8 acc = {0,0,0,0,0,0,0,0};
  #pragma unroll
  for (int kk = 0; kk < 8; ++kk){
    Q2 aq, bq;
    const bf16_t* ap = arow + kk*32 + half*8;      // K chunk [.. , ..+8) and [+16, +24)
    aq.a = *(const uint4*)ap;
    aq.b = *(const uint4*)(ap + 16);
    const bf16_t* bp = wp + (((size_t)(kk*16 + nt))*32 + lane)*16;
    bq.a = *(const uint4*)bp;
    bq.b = *(const uint4*)(bp + 8);
    acc = __builtin_amdgcn_wmma_f32_16x16x32_bf16(false, __builtin_bit_cast(bf16x16, aq),
                                                  false, __builtin_bit_cast(bf16x16, bq),
                                                  (short)0, acc, false, false);
  }
  int col = nt*16 + r;
  #pragma unroll
  for (int i = 0; i < 8; ++i)
    outp[(size_t)(mt*16 + half*8 + i)*D_ + col] = acc[i];
}

// ---------------- S1: per-(b,k) slot prep ----------------
__global__ void k_slotprep(const float* __restrict__ sf, const float* __restrict__ nsg,
                           const float* __restrict__ nsb, const float* __restrict__ wq,
                           const float* __restrict__ gpw, const float* __restrict__ gpb,
                           const float* __restrict__ gew2, const float* __restrict__ geb2,
                           float* __restrict__ params, float* __restrict__ wq2,
                           float* __restrict__ qb2, int writeRel, float* __restrict__ relout){
  __shared__ float xn[256], qsh[256], buf[256];
  int bk = blockIdx.x, t = threadIdx.x;
  const float* sfp = sf + (size_t)bk*260;
  float s = sfp[t];
  float mean = blockSum256(s, buf) * (1.f/256.f);
  float d = s - mean;
  float var = blockSum256(d*d, buf) * (1.f/256.f);
  xn[t] = d * rsqrtf(var + LNEPS_) * nsg[t] + nsb[t];
  __syncthreads();
  float q = 0.f;
  for (int c = 0; c < 256; ++c) q += xn[c] * wq[(size_t)c*256 + t];
  qsh[t] = q; __syncthreads();
  for (int cc = t; cc < 512; cc += 256){
    float a = 0.f;
    for (int dd = 0; dd < 256; ++dd) a += gew2[(size_t)cc*256 + dd] * qsh[dd];
    wq2[(size_t)bk*512 + cc] = a;
  }
  float pq = blockSum256(q * geb2[t], buf);
  if (t == 0) qb2[bk] = pq;
  float px = fminf(fmaxf(sfp[256], -1.f), 1.f);
  float py = fminf(fmaxf(sfp[257], -1.f), 1.f);
  float sx = fminf(fmaxf(sfp[258], 1e-3f), 2.f);
  float sy = fminf(fmaxf(sfp[259], 1e-3f), 2.f);
  float w0 = gpw[t]       / (sx * SCALE_);
  float w1 = gpw[256 + t] / (sy * SCALE_);
  float* pp = params + (size_t)bk*768;
  pp[t] = w0; pp[256 + t] = w1; pp[512 + t] = gpb[t] - px*w0 - py*w1;
  if (writeRel){
    for (int n = t; n < N_; n += 256){
      relout[((size_t)bk*N_ + n)*2 + 0] = (gridX(n) - px) / (sx * SCALE_);
      relout[((size_t)bk*N_ + n)*2 + 1] = (gridY(n) - py) / (sy * SCALE_);
    }
  }
}

// ---------------- S2K: fused LN + GEMM(256->512) + dot(wq2) -> dots ----------------
__global__ __launch_bounds__(128) void k_relk_dots(
    const float* __restrict__ kbuf, const bf16_t* __restrict__ w1P,
    const float* __restrict__ params, const float* __restrict__ wq2,
    const float* __restrict__ qb2, const float* __restrict__ lnG,
    const float* __restrict__ lnB, const float* __restrict__ b1,
    float* __restrict__ dots){
  __shared__ alignas(16) bf16_t At[4][16][256];
  __shared__ float p0s[256], p1s[256], pbs[256], lgs[256], lbs[256], b1s[512], wqs[512];
  int bk = blockIdx.x >> 4, seg = blockIdx.x & 15;
  int b  = bk >> 3;
  int t = threadIdx.x, wv = t >> 5, lane = t & 31, half = lane >> 4, r = lane & 15;
  const float* pp = params + (size_t)bk*768;
  for (int i = t; i < 256; i += 128){
    p0s[i] = pp[i]; p1s[i] = pp[256+i]; pbs[i] = pp[512+i];
    lgs[i] = lnG[i]; lbs[i] = lnB[i];
  }
  for (int i = t; i < 512; i += 128){ b1s[i] = b1[i]; wqs[i] = wq2[(size_t)bk*512 + i]; }
  __syncthreads();
  int n = seg*64 + wv*16 + r;
  const float* krow = kbuf + ((size_t)b*N_ + n)*D_;
  float gxn = gridX(n), gyn = gridY(n);
  int c0 = half*128;
  float s1 = 0.f, s2 = 0.f;
  for (int c = c0; c < c0+128; ++c){
    float tv = krow[c] + gxn*p0s[c] + gyn*p1s[c] + pbs[c];
    s1 += tv; s2 += tv*tv;
  }
  s1 += __shfl_xor(s1, 16, 32); s2 += __shfl_xor(s2, 16, 32);
  float mean = s1*(1.f/256.f);
  float rstd = rsqrtf(s2*(1.f/256.f) - mean*mean + LNEPS_);
  for (int c = c0; c < c0+128; ++c){
    float tv = krow[c] + gxn*p0s[c] + gyn*p1s[c] + pbs[c];
    At[wv][r][c] = (bf16_t)((tv - mean)*rstd*lgs[c] + lbs[c]);
  }
  __syncthreads();
  float dAcc[8] = {0,0,0,0,0,0,0,0};
  for (int ct = 0; ct < 32; ++ct){
    f32x8 acc = {0,0,0,0,0,0,0,0};
    #pragma unroll
    for (int kk = 0; kk < 8; ++kk){
      Q2 aq, bq;
      const bf16_t* ap = &At[wv][r][kk*32 + half*8];
      aq.a = *(const uint4*)ap;
      aq.b = *(const uint4*)(ap + 16);
      const bf16_t* bp = w1P + (((size_t)(kk*32 + ct))*32 + lane)*16;
      bq.a = *(const uint4*)bp;
      bq.b = *(const uint4*)(bp + 8);
      acc = __builtin_amdgcn_wmma_f32_16x16x32_bf16(false, __builtin_bit_cast(bf16x16, aq),
                                                    false, __builtin_bit_cast(bf16x16, bq),
                                                    (short)0, acc, false, false);
    }
    int col = ct*16 + r;
    float bv = b1s[col], wv2 = wqs[col];
    #pragma unroll
    for (int i = 0; i < 8; ++i){ float h = acc[i] + bv; h = h > 0.f ? h : 0.f; dAcc[i] += h * wv2; }
  }
  #pragma unroll
  for (int i = 0; i < 8; ++i){
    float v = dAcc[i];
    v += __shfl_xor(v, 1, 32); v += __shfl_xor(v, 2, 32);
    v += __shfl_xor(v, 4, 32); v += __shfl_xor(v, 8, 32);
    dAcc[i] = v;
  }
  if (r == 0){
    float qb = qb2[bk];
    #pragma unroll
    for (int i = 0; i < 8; ++i){
      int nn = seg*64 + wv*16 + half*8 + i;
      dots[(size_t)bk*N_ + nn] = (dAcc[i] + qb) * 0.0625f;   // D^-0.5
    }
  }
}

// ---------------- S2V: fused LN + GEMM(256->512) -> h1v (bf16, fragment layout) ----------------
__global__ __launch_bounds__(128) void k_relv_h1(
    const float* __restrict__ vbuf, const bf16_t* __restrict__ w1P,
    const float* __restrict__ params, const float* __restrict__ lnG,
    const float* __restrict__ lnB, const float* __restrict__ b1,
    bf16_t* __restrict__ h1v){
  __shared__ alignas(16) bf16_t At[4][16][256];
  __shared__ float p0s[256], p1s[256], pbs[256], lgs[256], lbs[256], b1s[512];
  int bk = blockIdx.x >> 4, seg = blockIdx.x & 15;
  int b  = bk >> 3;
  int t = threadIdx.x, wv = t >> 5, lane = t & 31, half = lane >> 4, r = lane & 15;
  const float* pp = params + (size_t)bk*768;
  for (int i = t; i < 256; i += 128){
    p0s[i] = pp[i]; p1s[i] = pp[256+i]; pbs[i] = pp[512+i];
    lgs[i] = lnG[i]; lbs[i] = lnB[i];
  }
  for (int i = t; i < 512; i += 128) b1s[i] = b1[i];
  __syncthreads();
  int n = seg*64 + wv*16 + r;
  const float* vrow = vbuf + ((size_t)b*N_ + n)*D_;
  float gxn = gridX(n), gyn = gridY(n);
  int c0 = half*128;
  float s1 = 0.f, s2 = 0.f;
  for (int c = c0; c < c0+128; ++c){
    float tv = vrow[c] + gxn*p0s[c] + gyn*p1s[c] + pbs[c];
    s1 += tv; s2 += tv*tv;
  }
  s1 += __shfl_xor(s1, 16, 32); s2 += __shfl_xor(s2, 16, 32);
  float mean = s1*(1.f/256.f);
  float rstd = rsqrtf(s2*(1.f/256.f) - mean*mean + LNEPS_);
  for (int c = c0; c < c0+128; ++c){
    float tv = vrow[c] + gxn*p0s[c] + gyn*p1s[c] + pbs[c];
    At[wv][r][c] = (bf16_t)((tv - mean)*rstd*lgs[c] + lbs[c]);
  }
  __syncthreads();
  int ntile = seg*4 + wv;
  for (int ct = 0; ct < 32; ++ct){
    f32x8 acc = {0,0,0,0,0,0,0,0};
    #pragma unroll
    for (int kk = 0; kk < 8; ++kk){
      Q2 aq, bq;
      const bf16_t* ap = &At[wv][r][kk*32 + half*8];
      aq.a = *(const uint4*)ap;
      aq.b = *(const uint4*)(ap + 16);
      const bf16_t* bp = w1P + (((size_t)(kk*32 + ct))*32 + lane)*16;
      bq.a = *(const uint4*)bp;
      bq.b = *(const uint4*)(bp + 8);
      acc = __builtin_amdgcn_wmma_f32_16x16x32_bf16(false, __builtin_bit_cast(bf16x16, aq),
                                                    false, __builtin_bit_cast(bf16x16, bq),
                                                    (short)0, acc, false, false);
    }
    float bv = b1s[ct*16 + r];
    BF8 o;
    #pragma unroll
    for (int i = 0; i < 8; ++i){ float h = acc[i] + bv; o.e[i] = (bf16_t)(h > 0.f ? h : 0.f); }
    *(uint4*)(h1v + ((((size_t)bk*64 + ntile)*32 + ct)*32 + lane)*8) = __builtin_bit_cast(uint4, o);
  }
}

// ---------------- S3a: softmax over K (axis=1) ----------------
__global__ void k_softmaxK(const float* __restrict__ dots, float* __restrict__ attn0){
  int idx = blockIdx.x*256 + threadIdx.x;        // 0..8191 = (b,n)
  int b = idx >> 10, n = idx & 1023;
  float v[8]; float mx = -1e30f;
  #pragma unroll
  for (int kk = 0; kk < 8; ++kk){ v[kk] = dots[((size_t)(b*8 + kk))*N_ + n]; mx = fmaxf(mx, v[kk]); }
  float s = 0.f;
  #pragma unroll
  for (int kk = 0; kk < 8; ++kk){ v[kk] = expf(v[kk] - mx); s += v[kk]; }
  float inv = 1.f / s;
  #pragma unroll
  for (int kk = 0; kk < 8; ++kk) attn0[((size_t)(b*8 + kk))*N_ + n] = v[kk] * inv;
}

// ---------------- S3b: closed-form pos_n / scl_n / attn normalizer ----------------
__global__ void k_moments(const float* __restrict__ attn0, float* __restrict__ stats){
  __shared__ float buf[256];
  int bk = blockIdx.x, t = threadIdx.x;
  float s0 = 0.f, sx = 0.f, sy = 0.f, sxx = 0.f, syy = 0.f;
  for (int n = t; n < N_; n += 256){
    float a = attn0[(size_t)bk*N_ + n];
    float gx = gridX(n), gy = gridY(n);
    s0 += a; sx += a*gx; sy += a*gy; sxx += a*gx*gx; syy += a*gy*gy;
  }
  s0 = blockSum256(s0, buf); sx = blockSum256(sx, buf); sy = blockSum256(sy, buf);
  sxx = blockSum256(sxx, buf); syy = blockSum256(syy, buf);
  if (t == 0){
    float G1 = 0.f, G2 = 0.f;
    for (int i = 0; i < 32; ++i){ float xv = -1.f + (2.f/31.f)*i; G1 += xv; G2 += xv*xv; }
    G1 *= 32.f; G2 *= 32.f;
    float px = sx, py = sy;
    float vx = sxx - 2.f*px*sx + px*px*s0 + EPS_*(G2 - 2.f*px*G1 + (float)N_*px*px);
    float vy = syy - 2.f*py*sy + py*py*s0 + EPS_*(G2 - 2.f*py*G1 + (float)N_*py*py);
    float cx = fminf(fmaxf(sqrtf(fmaxf(vx, 0.f)), 1e-3f), 2.f);
    float cy = fminf(fmaxf(sqrtf(fmaxf(vy, 0.f)), 1e-3f), 2.f);
    float* sp = stats + bk*8;
    sp[0] = px; sp[1] = py; sp[2] = cx; sp[3] = cy;
    sp[4] = 1.f / (s0 + (float)N_ * EPS_);
  }
}

// ---------------- S4: acc512[c] = sum_n attn[n] * h1v[n,c] ----------------
__global__ __launch_bounds__(1024) void k_updAcc(const bf16_t* __restrict__ h1v,
                          const float* __restrict__ attn0, const float* __restrict__ stats,
                          float* __restrict__ acc512){
  int bk = blockIdx.x, t = threadIdx.x;
  int ct = t >> 5, lane = t & 31, half = lane >> 4, col = ct*16 + (lane & 15);
  float invd = stats[bk*8 + 4];
  float acc = 0.f;
  for (int nt = 0; nt < 64; ++nt){
    BF8 u = __builtin_bit_cast(BF8, *(const uint4*)(h1v + ((((size_t)bk*64 + nt)*32 + ct)*32 + lane)*8));
    const float* ap = attn0 + (size_t)bk*N_ + nt*16 + half*8;
    #pragma unroll
    for (int i = 0; i < 8; ++i) acc += (float)u.e[i] * ((ap[i] + EPS_) * invd);
  }
  acc += __shfl_xor(acc, 16, 32);
  if (half == 0) acc512[(size_t)bk*512 + col] = acc;
}

// ---------------- S5: upd = acc512@W2+b2, GRU, repack slots_full ----------------
__global__ void k_gru(const float* __restrict__ sfIn, const float* __restrict__ acc512,
                      const float* __restrict__ gew2, const float* __restrict__ geb2,
                      const float* __restrict__ wih, const float* __restrict__ whh,
                      const float* __restrict__ bih, const float* __restrict__ bhh,
                      const float* __restrict__ stats, int doUpd, float* __restrict__ sfOut){
  __shared__ float upd[256], sl[256];
  int bk = blockIdx.x, t = threadIdx.x;
  sl[t] = sfIn[(size_t)bk*260 + t];
  if (doUpd){
    float s = geb2[t];
    const float* av = acc512 + (size_t)bk*512;
    for (int c = 0; c < 512; ++c) s += av[c] * gew2[(size_t)c*256 + t];
    upd[t] = s;
  }
  __syncthreads();
  float out;
  if (doUpd){
    float gr = bih[t], gz = bih[256+t], gn = bih[512+t];
    float hr = bhh[t], hz = bhh[256+t], hn = bhh[512+t];
    const float* wr = wih + (size_t)t*256;
    const float* wz = wih + (size_t)(t+256)*256;
    const float* wn = wih + (size_t)(t+512)*256;
    const float* vr = whh + (size_t)t*256;
    const float* vz = whh + (size_t)(t+256)*256;
    const float* vn = whh + (size_t)(t+512)*256;
    for (int d2 = 0; d2 < 256; ++d2){
      float u = upd[d2], s = sl[d2];
      gr += u*wr[d2]; gz += u*wz[d2]; gn += u*wn[d2];
      hr += s*vr[d2]; hz += s*vz[d2]; hn += s*vn[d2];
    }
    float rr = 1.f/(1.f + expf(-(gr + hr)));
    float zz = 1.f/(1.f + expf(-(gz + hz)));
    float nn = tanhf(gn + rr*hn);
    out = (1.f - zz)*nn + zz*sl[t];
  } else out = sl[t];
  sfOut[(size_t)bk*260 + t] = out;
  if (t < 4) sfOut[(size_t)bk*260 + 256 + t] = stats[bk*8 + t];
}

// ---------------- utility copies ----------------
__global__ void k_copy(const float* __restrict__ src, float* __restrict__ dst, int n){
  int i = blockIdx.x*256 + threadIdx.x;
  if (i < n) dst[i] = src[i];
}
__global__ void k_gather(const float* __restrict__ sf, const float* __restrict__ attn0,
                         const float* __restrict__ rel, float* __restrict__ out){
  int i = blockIdx.x*256 + threadIdx.x;
  if (i < 16640)            out[i] = sf[i];
  else if (i < 82176)       out[i] = attn0[i - 16640];
  else if (i < 213248)      out[i] = rel[i - 82176];
}

extern "C" void kernel_launch(void* const* d_in, const int* in_sizes, int n_in,
                              void* d_out, int out_size, void* d_ws, size_t ws_size,
                              hipStream_t stream){
  (void)in_sizes; (void)n_in; (void)out_size; (void)ws_size;
  const float* inputs = (const float*)d_in[0];
  const float* cond   = (const float*)d_in[1];
  const float* ni_g   = (const float*)d_in[2];
  const float* ni_b   = (const float*)d_in[3];
  const float* ns_g   = (const float*)d_in[4];
  const float* ns_b   = (const float*)d_in[5];
  const float* wq     = (const float*)d_in[6];
  const float* wk     = (const float*)d_in[7];
  const float* wv     = (const float*)d_in[8];
  const float* gp_w   = (const float*)d_in[9];
  const float* gp_b   = (const float*)d_in[10];
  const float* lnG    = (const float*)d_in[11];
  const float* lnB    = (const float*)d_in[12];
  const float* w1     = (const float*)d_in[13];
  const float* b1     = (const float*)d_in[14];
  const float* w2     = (const float*)d_in[15];
  const float* b2     = (const float*)d_in[16];
  const float* wih    = (const float*)d_in[17];
  const float* whh    = (const float*)d_in[18];
  const float* bih    = (const float*)d_in[19];
  const float* bhh    = (const float*)d_in[20];

  char* base = (char*)d_ws;
  size_t off = 0;
  auto alloc = [&](size_t bytes)->char*{
    char* p = base + off; off = (off + bytes + 255) & ~(size_t)255; return p;
  };
  bf16_t* xq    = (bf16_t*)alloc(8192ull*256*2);
  bf16_t* wkP   = (bf16_t*)alloc(131072);
  bf16_t* wvP   = (bf16_t*)alloc(131072);
  bf16_t* w1P   = (bf16_t*)alloc(262144);
  float*  kbuf  = (float*) alloc(8192ull*256*4);
  float*  vbuf  = (float*) alloc(8192ull*256*4);
  float*  sf0   = (float*) alloc(66560);
  float*  sf1   = (float*) alloc(66560);
  float*  params= (float*) alloc(196608);
  float*  wq2   = (float*) alloc(131072);
  float*  qb2   = (float*) alloc(256);
  float*  dots  = (float*) alloc(262144);
  float*  attn0 = (float*) alloc(262144);
  float*  relb  = (float*) alloc(524288);
  float*  stats = (float*) alloc(2048);
  float*  acc512= (float*) alloc(131072);
  bf16_t* h1v   = (bf16_t*)alloc(33554432ull*2);

  k_ln_inputs<<<8192, 256, 0, stream>>>(inputs, ni_g, ni_b, xq);
  k_packB<<<128, 32, 0, stream>>>(wk, wkP, 16);
  k_packB<<<128, 32, 0, stream>>>(wv, wvP, 16);
  k_packB<<<256, 32, 0, stream>>>(w1, w1P, 32);
  k_proj<<<4096, 128, 0, stream>>>(xq, wkP, wvP, kbuf, vbuf);
  k_copy<<<65, 256, 0, stream>>>(cond, sf0, 16640);

  float* cur = sf0; float* nxt = sf1;
  for (int it = 0; it < 4; ++it){
    int upd = (it < 3) ? 1 : 0;
    int last = (it == 3) ? 1 : 0;
    k_slotprep<<<64, 256, 0, stream>>>(cur, ns_g, ns_b, wq, gp_w, gp_b, w2, b2,
                                       params, wq2, qb2, last, relb);
    k_relk_dots<<<1024, 128, 0, stream>>>(kbuf, w1P, params, wq2, qb2, lnG, lnB, b1, dots);
    if (upd) k_relv_h1<<<1024, 128, 0, stream>>>(vbuf, w1P, params, lnG, lnB, b1, h1v);
    k_softmaxK<<<32, 256, 0, stream>>>(dots, attn0);
    k_moments<<<64, 256, 0, stream>>>(attn0, stats);
    if (upd) k_updAcc<<<64, 1024, 0, stream>>>(h1v, attn0, stats, acc512);
    k_gru<<<64, 256, 0, stream>>>(cur, acc512, w2, b2, wih, whh, bih, bhh, stats, upd, nxt);
    float* tmp = cur; cur = nxt; nxt = tmp;
  }
  k_gather<<<833, 256, 0, stream>>>(cur, attn0, relb, (float*)d_out);
}
